// ResNet101_PsROI_Head_85306640433492
// MI455X (gfx1250) — matrix operand order repaired and checked by reading, not compiled, not run on hardware
//
#include <hip/hip_runtime.h>

typedef unsigned short ushort_t;
typedef __attribute__((ext_vector_type(16))) __bf16 v16bf;
typedef __attribute__((ext_vector_type(8)))  float  v8f;
typedef __attribute__((__vector_size__(4 * sizeof(int)))) int i32x4_t;

#define K_DIM      1024
#define P_DIM      4096   // 64*64 spatial positions
#define H_DIM      64
#define W_DIM      64
#define M_SCORE    1029   // 21*7*7
#define M_LOC      392    // 2*4*7*7
#define KBINS      7
#define SCALE_F    (1.0f/16.0f)

#if defined(__has_builtin)
#if __has_builtin(__builtin_amdgcn_global_load_async_to_lds_b128) && \
    __has_builtin(__builtin_amdgcn_s_wait_asynccnt)
#define USE_ASYNC_LDS 1
#endif
#endif

#ifdef USE_ASYNC_LDS
// builtin signature (from hipcc diagnostic): param0 = int4 addrspace(1)*, then lds ptr, imm offset, cpol
#define GAS1(p) ((__attribute__((address_space(1))) i32x4_t*)(p))
#define LAS3(p) ((__attribute__((address_space(3))) i32x4_t*)(p))
#endif

// ---- workspace layout (bytes) ----
static constexpr size_t OFF_XT  = 0;                                    // xT bf16 [4096][1024]
static constexpr size_t OFF_WS  = OFF_XT + (size_t)P_DIM * K_DIM * 2;   // Wscore bf16
static constexpr size_t OFF_WL  = OFF_WS + (size_t)M_SCORE * K_DIM * 2; // Wloc bf16
static constexpr size_t OFF_SM  = OFF_WL + (size_t)M_LOC  * K_DIM * 2;  // score map f32
static constexpr size_t OFF_LM  = OFF_SM + (size_t)M_SCORE * P_DIM * 4; // loc map f32

__device__ __forceinline__ ushort_t f2bf(float f) {
    unsigned int u = __float_as_uint(f);
    unsigned int r = u + 0x7fffu + ((u >> 16) & 1u);   // round-to-nearest-even
    return (ushort_t)(r >> 16);
}

// ---------------- x transpose + f32->bf16:  x[c][p] -> xT[p][c] ----------------
__global__ void transpose_cvt_kernel(const float* __restrict__ x,
                                     ushort_t* __restrict__ xT) {
    __shared__ float tile[32][33];
    int bx = blockIdx.x;             // p tile
    int by = blockIdx.y;             // c tile
    #pragma unroll
    for (int j = 0; j < 4; ++j) {
        int c = by * 32 + threadIdx.y + j * 8;
        int p = bx * 32 + threadIdx.x;
        tile[threadIdx.y + j * 8][threadIdx.x] = x[(size_t)c * P_DIM + p];
    }
    __syncthreads();
    #pragma unroll
    for (int j = 0; j < 4; ++j) {
        int p = bx * 32 + threadIdx.y + j * 8;
        int c = by * 32 + threadIdx.x;
        xT[(size_t)p * K_DIM + c] = f2bf(tile[threadIdx.x][threadIdx.y + j * 8]);
    }
}

// ---------------- elementwise f32 -> bf16 ----------------
__global__ void cvt_bf16_kernel(const float* __restrict__ in,
                                ushort_t* __restrict__ out, int n) {
    for (int i = blockIdx.x * blockDim.x + threadIdx.x; i < n; i += gridDim.x * blockDim.x)
        out[i] = f2bf(in[i]);
}

// ---------------- WMMA GEMM:  C[o][p] = sum_c A[o][c] * B[p][c] ----------------
// A: [M][1024] bf16 row-major (weights).  B: [4096][1024] bf16 row-major (xT).
// Block = 256 threads (8 waves). C tile = 128(M) x 64(N). K-step = 32.
// Double-buffered LDS; async global->LDS copies (ASYNCcnt) when available.
union FragBF { v16bf v; uint4 q[2]; };

#define A_STRIDE 40   // ushorts per A row in LDS (80B; keeps 16B-aligned frags)
#define B_STRIDE 40
#define A_TILE   (128 * A_STRIDE)
#define B_TILE   (64 * B_STRIDE)

__global__ void __launch_bounds__(256)
gemm_wmma_kernel(const ushort_t* __restrict__ A,
                 const ushort_t* __restrict__ B,
                 float* __restrict__ C, int M) {
    __shared__ __align__(16) ushort_t As[2 * A_TILE];
    __shared__ __align__(16) ushort_t Bs[2 * B_TILE];

    const int tid  = threadIdx.x;
    const int lane = tid & 31;
    const int wid  = tid >> 5;
    const int half = lane >> 4;
    const int l16  = lane & 15;

    const int nbase = blockIdx.x * 64;
    const int mbase = blockIdx.y * 128;

    // cooperative-load indexing (each thread: 32B of A, 16B of B per K-step)
    const int a_row = tid >> 1, a_half = tid & 1;
    const int b_row = tid >> 2, b_seg  = tid & 3;
    int a_grow = mbase + a_row;
    if (a_grow >= M) a_grow = M - 1;          // clamp: stays in-bounds, rows >= M never stored
    const ushort_t* srcA = A + (size_t)a_grow * K_DIM + a_half * 16;
    const ushort_t* srcB = B + (size_t)(nbase + b_row) * K_DIM + b_seg * 8;
    ushort_t* dstA = &As[a_row * A_STRIDE + a_half * 16];
    ushort_t* dstB = &Bs[b_row * B_STRIDE + b_seg * 8];

    v8f acc[4];
    #pragma unroll
    for (int i = 0; i < 4; ++i)
        #pragma unroll
        for (int j = 0; j < 8; ++j) acc[i][j] = 0.0f;

#ifdef USE_ASYNC_LDS
    // ---- async double-buffered pipeline ----
    // prologue: stage tile 0 into buffer 0
    __builtin_amdgcn_global_load_async_to_lds_b128(GAS1(srcA),      LAS3(dstA),      0, 0);
    __builtin_amdgcn_global_load_async_to_lds_b128(GAS1(srcA + 8),  LAS3(dstA + 8),  0, 0);
    __builtin_amdgcn_global_load_async_to_lds_b128(GAS1(srcB),      LAS3(dstB),      0, 0);

    for (int it = 0; it < K_DIM / 32; ++it) {
        const int cur = it & 1;
        const int k0  = it * 32;
        __builtin_amdgcn_s_wait_asynccnt(0);  // current tile landed in LDS
        __syncthreads();                      // visible to all waves; prev buffer free
        if (it + 1 < K_DIM / 32) {            // overlap next-tile copy with compute
            const int nk = k0 + 32;
            const int nb = (cur ^ 1);
            __builtin_amdgcn_global_load_async_to_lds_b128(
                GAS1(srcA + nk),     LAS3(dstA + nb * A_TILE),     0, 0);
            __builtin_amdgcn_global_load_async_to_lds_b128(
                GAS1(srcA + nk + 8), LAS3(dstA + nb * A_TILE + 8), 0, 0);
            __builtin_amdgcn_global_load_async_to_lds_b128(
                GAS1(srcB + nk),     LAS3(dstB + nb * B_TILE),     0, 0);
            if (nk + 32 < K_DIM) __builtin_prefetch(srcB + nk + 32, 0, 0);
        }
        const ushort_t* asb = &As[cur * A_TILE];
        const ushort_t* bsb = &Bs[cur * B_TILE];
#else
    // ---- synchronous fallback: single-buffer staging through VGPRs ----
    for (int it = 0; it < K_DIM / 32; ++it) {
        const int k0 = it * 32;
        if (k0) __syncthreads();
        {
            uint4* dst = reinterpret_cast<uint4*>(dstA);
            const uint4* s = reinterpret_cast<const uint4*>(srcA + k0);
            dst[0] = s[0]; dst[1] = s[1];
            uint4* db = reinterpret_cast<uint4*>(dstB);
            db[0] = *reinterpret_cast<const uint4*>(srcB + k0);
        }
        if (k0 + 32 < K_DIM) {
            __builtin_prefetch(srcA + k0 + 32, 0, 0);
            __builtin_prefetch(srcB + k0 + 32, 0, 0);
        }
        __syncthreads();
        const ushort_t* asb = &As[0];
        const ushort_t* bsb = &Bs[0];
#endif
        // A fragment: lane m = l16; e0..7 = K half*8+0..7, e8..15 = K 16+half*8+0..7
        FragBF fa;
        {
            const int mrow = wid * 16 + l16;
            fa.q[0] = *reinterpret_cast<const uint4*>(&asb[mrow * A_STRIDE + half * 8]);
            fa.q[1] = *reinterpret_cast<const uint4*>(&asb[mrow * A_STRIDE + 16 + half * 8]);
        }
        // preload all four B fragments (distinct regs -> loads overlap WMMA issue)
        FragBF fb[4];
        #pragma unroll
        for (int ns = 0; ns < 4; ++ns) {
            const int bidx = (ns * 16 + l16) * B_STRIDE + half * 8;
            fb[ns].q[0] = *reinterpret_cast<const uint4*>(&bsb[bidx]);
            fb[ns].q[1] = *reinterpret_cast<const uint4*>(&bsb[bidx + 16]);
        }
        #pragma unroll
        for (int ns = 0; ns < 4; ++ns)
            acc[ns] = __builtin_amdgcn_wmma_f32_16x16x32_bf16(
                false, fa.v, false, fb[ns].v, (short)0, acc[ns], false, false);
    }

    // store: VGPR i -> M = i + 8*(lane>=16), N = l16
    #pragma unroll
    for (int ns = 0; ns < 4; ++ns) {
        const int col   = nbase + ns * 16 + l16;
        const int rbase = mbase + wid * 16 + half * 8;
        #pragma unroll
        for (int i = 0; i < 8; ++i) {
            int row = rbase + i;
            if (row < M) C[(size_t)row * P_DIM + col] = acc[ns][i];
        }
    }
}

// ---------------- PS-ROI pool + mean, one block per ROI ----------------
__global__ void __launch_bounds__(256)
psroi_kernel(const float* __restrict__ smap, const float* __restrict__ lmap,
             const float* __restrict__ rois,
             const float* __restrict__ bscore, const float* __restrict__ bloc,
             float* __restrict__ out) {
    __shared__ int hs[KBINS], he[KBINS], wss[KBINS], wes[KBINS];
    __shared__ float accS[21], accL[8];

    const int r   = blockIdx.x;
    const int tid = threadIdx.x;

    if (tid == 0) {
        float y0 = rois[r * 4 + 0], x0 = rois[r * 4 + 1];
        float y1 = rois[r * 4 + 2], x1 = rois[r * 4 + 3];
        float sh = rintf(y0) * SCALE_F, eh = (rintf(y1) + 1.0f) * SCALE_F;
        float sw = rintf(x0) * SCALE_F, ew = (rintf(x1) + 1.0f) * SCALE_F;
        float bh = fmaxf(eh - sh, 0.1f) * (1.0f / KBINS);
        float bw = fmaxf(ew - sw, 0.1f) * (1.0f / KBINS);
        #pragma unroll
        for (int p = 0; p < KBINS; ++p) {
            hs[p]  = (int)fminf(fmaxf(floorf((float)p * bh + sh), 0.0f), (float)H_DIM);
            he[p]  = (int)fminf(fmaxf(ceilf(((float)p + 1.0f) * bh + sh), 0.0f), (float)H_DIM);
            wss[p] = (int)fminf(fmaxf(floorf((float)p * bw + sw), 0.0f), (float)W_DIM);
            wes[p] = (int)fminf(fmaxf(ceilf(((float)p + 1.0f) * bw + sw), 0.0f), (float)W_DIM);
        }
    }
    if (tid < 21) accS[tid] = 0.0f;
    if (tid < 8)  accL[tid] = 0.0f;
    __syncthreads();

    for (int o = tid; o < M_SCORE + M_LOC; o += 256) {
        const bool is_score = (o < M_SCORE);
        const int  oo   = is_score ? o : (o - M_SCORE);
        const int  ctop = oo / 49;
        const int  bin  = oo % 49;
        const int  ph   = bin / 7, pw = bin % 7;
        const int  h0 = hs[ph], h1 = he[ph], w0 = wss[pw], w1 = wes[pw];
        const float area = (float)((h1 - h0) * (w1 - w0));
        if (area > 0.0f) {
            const float* mp = (is_score ? smap : lmap) + (size_t)oo * P_DIM;
            float s = 0.0f;
            for (int h = h0; h < h1; ++h) {
                const float* rowp = mp + h * W_DIM;
                for (int w = w0; w < w1; ++w) s += rowp[w];
            }
            float bias = is_score ? bscore[oo] : bloc[oo];
            float val  = s / area + bias;
            atomicAdd(is_score ? &accS[ctop] : &accL[ctop], val);
        }
    }
    __syncthreads();

    const float inv49 = 1.0f / 49.0f;
    if (tid < 8)  out[(size_t)r * 8 + tid] = accL[tid] * inv49;          // roi_locs
    if (tid < 21) out[8192 + (size_t)r * 21 + tid] = accS[tid] * inv49;  // roi_scores
}

extern "C" void kernel_launch(void* const* d_in, const int* in_sizes, int n_in,
                              void* d_out, int out_size, void* d_ws, size_t ws_size,
                              hipStream_t stream) {
    (void)in_sizes; (void)n_in; (void)out_size; (void)ws_size;
    const float* x      = (const float*)d_in[0];
    const float* rois   = (const float*)d_in[1];
    /* roi_indices (all zero, N=1) unused:   d_in[2] */
    const float* Wscore = (const float*)d_in[3];
    const float* bscore = (const float*)d_in[4];
    const float* Wloc   = (const float*)d_in[5];
    const float* bloc   = (const float*)d_in[6];
    float* out = (float*)d_out;

    char* ws = (char*)d_ws;
    ushort_t* xT   = (ushort_t*)(ws + OFF_XT);
    ushort_t* Wsbf = (ushort_t*)(ws + OFF_WS);
    ushort_t* Wlbf = (ushort_t*)(ws + OFF_WL);
    float*    smap = (float*)(ws + OFF_SM);
    float*    lmap = (float*)(ws + OFF_LM);

    transpose_cvt_kernel<<<dim3(P_DIM / 32, K_DIM / 32), dim3(32, 8), 0, stream>>>(x, xT);
    cvt_bf16_kernel<<<512, 256, 0, stream>>>(Wscore, Wsbf, M_SCORE * K_DIM);
    cvt_bf16_kernel<<<512, 256, 0, stream>>>(Wloc,   Wlbf, M_LOC  * K_DIM);

    gemm_wmma_kernel<<<dim3(P_DIM / 64, (M_SCORE + 127) / 128), 256, 0, stream>>>(Wsbf, xT, smap, M_SCORE);
    gemm_wmma_kernel<<<dim3(P_DIM / 64, (M_LOC  + 127) / 128), 256, 0, stream>>>(Wlbf, xT, lmap, M_LOC);

    psroi_kernel<<<1024, 256, 0, stream>>>(smap, lmap, rois, bscore, bloc, out);
}
